// TriangleSelfAttentionStartingNode_18373870092373
// MI455X (gfx1250) — matrix operand order repaired and checked by reading, not compiled, not run on hardware
//
#include <hip/hip_runtime.h>
#include <cstdint>

typedef __attribute__((ext_vector_type(16))) _Float16 v16h;
typedef __attribute__((ext_vector_type(8)))  _Float16 v8h;
typedef __attribute__((ext_vector_type(4)))  _Float16 v4h;
typedef __attribute__((ext_vector_type(8)))  float    v8f;
typedef __attribute__((ext_vector_type(4)))  float    v4f;
typedef __attribute__((ext_vector_type(4)))  int      v4i;

// problem sizes
static constexpr int NN   = 512;
static constexpr int DD   = 128;
static constexpr int HH   = 4;
static constexpr int NP   = NN * NN;     // 262144 spatial positions
static constexpr int FF   = 512;         // fused qkvg width (4*C)
static constexpr int KTOT = 16384;       // score contraction length (512 i' * 32 c)

// workspace offsets (bytes)
static constexpr size_t XN_OFF    = 0;                                   // f16 [NP][128]
static constexpr size_t Q_OFF     = XN_OFF   + (size_t)NP * DD * 2;      // f16 [4][16384][512]
static constexpr size_t K_OFF     = Q_OFF    + (size_t)HH * KTOT * NN * 2;
static constexpr size_t G_OFF     = K_OFF    + (size_t)HH * KTOT * NN * 2; // f16 [4][512][512][32]
static constexpr size_t BIAS_OFF  = G_OFF    + (size_t)HH * NP * 32 * 2;   // f32 [4][512][512]
static constexpr size_t SCORE_OFF = BIAS_OFF + (size_t)HH * NP * 4;        // f32 [4][512][512]
static constexpr size_t VSUM_OFF  = SCORE_OFF+ (size_t)HH * NP * 4;        // f32 [4][32]
static constexpr size_t WPF_OFF   = VSUM_OFF + 512;                        // f16 W_proj B-frags
static constexpr size_t WOF_OFF   = WPF_OFF  + (size_t)4 * 32 * 32 * 16 * 2;
static constexpr size_t SXN_OFF   = WOF_OFF  + (size_t)4 * 8 * 32 * 16 * 2; // f32 [256][128]

#define WMMA_F32_F16(a, b, c) \
  __builtin_amdgcn_wmma_f32_16x16x32_f16(false, (a), false, (b), (short)0, (c), false, false)

// ---- gfx1250 async-to-LDS path (builtin exists on this toolchain; the
//      parameters are v4i addrspace(1)* / v4i addrspace(3)* per diagnostics) --
#if __has_builtin(__builtin_amdgcn_global_load_async_to_lds_b128)
#define ATHENA_ASYNC_LDS 1
typedef __attribute__((address_space(1))) v4i as1_v4i;
typedef __attribute__((address_space(3))) v4i as3_v4i;
static __device__ __forceinline__ void async_copy_b128(const void* g, void* lds) {
  // (global src, lds dst, imm offset, imm cpol) -> global_load_async_to_lds_b128
  __builtin_amdgcn_global_load_async_to_lds_b128(
      (as1_v4i*)(unsigned long long)(uintptr_t)g,
      (as3_v4i*)(unsigned int)(uintptr_t)lds, 0, 0);
}
static __device__ __forceinline__ void wait_async_zero() {
#if __has_builtin(__builtin_amdgcn_s_wait_asynccnt)
  __builtin_amdgcn_s_wait_asynccnt(0);
#else
  asm volatile("s_wait_asynccnt 0" ::: "memory");
#endif
}
#endif

static __device__ __forceinline__ v16h cat16(v8h lo, v8h hi) {
  v16h r;
#pragma unroll
  for (int s = 0; s < 8; ++s) { r[s] = lo[s]; r[s + 8] = hi[s]; }
  return r;
}

static __device__ __forceinline__ v8f zero_v8f() {
  v8f z;
#pragma unroll
  for (int i = 0; i < 8; ++i) z[i] = 0.f;
  return z;
}

// ---------------------------------------------------------------------------
// K0: pack W_proj / W_out into per-lane WMMA B-fragment order (f16).
//     B 32x16 tile: lane = n + 16*((k>>3)&1), slot = ((k>>4)<<3) | (k&7).
// ---------------------------------------------------------------------------
__global__ __launch_bounds__(256) void k_pack(const float* __restrict__ wproj,
                                              const float* __restrict__ wout,
                                              _Float16* __restrict__ wpf,
                                              _Float16* __restrict__ wof) {
  const int tid = blockIdx.x * 256 + threadIdx.x;
  if (tid < 4 * 32 * 32 * 16) {
    const int slot = tid & 15, lane = (tid >> 4) & 31, nt = (tid >> 9) & 31, kt = tid >> 14;
    const int kk = ((slot & 8) << 1) | ((lane >> 4) << 3) | (slot & 7);
    wpf[tid] = (_Float16)wproj[(kt * 32 + kk) * FF + nt * 16 + (lane & 15)];
  } else if (tid - 65536 < 4 * 8 * 32 * 16) {
    const int t2 = tid - 65536;
    const int slot = t2 & 15, lane = (t2 >> 4) & 31, nt = (t2 >> 9) & 7, kt = t2 >> 12;
    const int kk = ((slot & 8) << 1) | ((lane >> 4) << 3) | (slot & 7);
    wof[t2] = (_Float16)wout[(kt * 32 + kk) * DD + nt * 16 + (lane & 15)];
  }
}

// ---------------------------------------------------------------------------
// K1: LayerNorm (one wave per position, wave32 shuffle reductions),
//     writes xn (f16) and bias[h][i][j] = xn . W_bias[:,h] (f32).
// ---------------------------------------------------------------------------
__global__ __launch_bounds__(256) void k_ln(const float* __restrict__ x,
                                            const float* __restrict__ lnw,
                                            const float* __restrict__ lnb,
                                            const float* __restrict__ wbias,
                                            _Float16* __restrict__ xn,
                                            float* __restrict__ bias) {
  const int gid  = blockIdx.x * blockDim.x + threadIdx.x;
  const int pos  = gid >> 5;
  const int lane = threadIdx.x & 31;
  if (pos >= NP) return;
  const v4f xv = *(const v4f*)(x + (size_t)pos * DD + lane * 4);
  float s = 0.f, s2 = 0.f;
#pragma unroll
  for (int i = 0; i < 4; ++i) { s += xv[i]; s2 += xv[i] * xv[i]; }
#pragma unroll
  for (int m = 16; m > 0; m >>= 1) {
    s  += __shfl_xor(s,  m, 32);
    s2 += __shfl_xor(s2, m, 32);
  }
  const float mu  = s * (1.f / 128.f);
  const float var = s2 * (1.f / 128.f) - mu * mu;   // biased variance, matches jnp.var
  const float inv = rsqrtf(var + 1e-5f);
  const v4f wv = *(const v4f*)(lnw + lane * 4);
  const v4f bv = *(const v4f*)(lnb + lane * 4);
  float xr[4];
  v4h hv;
#pragma unroll
  for (int i = 0; i < 4; ++i) {
    xr[i] = (xv[i] - mu) * inv * wv[i] + bv[i];
    hv[i] = (_Float16)xr[i];
  }
  *(v4h*)(xn + (size_t)pos * DD + lane * 4) = hv;
  const int d0 = lane * 4;
#pragma unroll
  for (int h = 0; h < HH; ++h) {
    float a = 0.f;
#pragma unroll
    for (int i = 0; i < 4; ++i) a += xr[i] * wbias[(d0 + i) * HH + h];
#pragma unroll
    for (int m = 16; m > 0; m >>= 1) a += __shfl_xor(a, m, 32);
    if (lane == 0) bias[(size_t)h * NP + pos] = a;
  }
}

// ---------------------------------------------------------------------------
// K2: proj GEMM  [262144 x 128] @ [128 x 512]  via WMMA f16, then scatter
//     through the torch reshape into Q[h][i'*32+c][j'], K[...], G[h][i'][j'][c].
// ---------------------------------------------------------------------------
__global__ __launch_bounds__(256) void k_proj(const _Float16* __restrict__ xn,
                                              const _Float16* __restrict__ wpf,
                                              _Float16* __restrict__ Qm,
                                              _Float16* __restrict__ Km,
                                              _Float16* __restrict__ Gm) {
  const int w = threadIdx.x >> 5, lane = threadIdx.x & 31;
  const int lm = lane & 15, hi = lane >> 4;
  const int mtile = blockIdx.x * 2 + (w >> 2);   // 16384 row tiles of 16
  const int ntg = (w & 3) * 8;                   // this wave's 8 col subtiles
  const int m0 = mtile * 16;
  v8f acc[8];
#pragma unroll
  for (int j = 0; j < 8; ++j) acc[j] = zero_v8f();
  const _Float16* arow = xn + (size_t)(m0 + lm) * DD + hi * 8;
#pragma unroll
  for (int kt = 0; kt < 4; ++kt) {
    const v8h a0 = *(const v8h*)(arow + kt * 32);
    const v8h a1 = *(const v8h*)(arow + kt * 32 + 16);
    const v16h af = cat16(a0, a1);
#pragma unroll
    for (int j = 0; j < 8; ++j) {
      const v16h bf = *(const v16h*)(wpf + ((size_t)((kt * 32 + ntg + j) * 32 + lane) << 4));
      acc[j] = WMMA_F32_F16(af, bf, acc[j]);
    }
  }
  // scatter: C/D layout lane<16 -> M=r, lane>=16 -> M=r+8; N=lane&15
#pragma unroll
  for (int j = 0; j < 8; ++j) {
    const int f = (ntg + j) * 16 + lm;
    const int cp = f & 127, fhi = f >> 7;
    const int chunk = cp >> 5, c = cp & 31;
    if (chunk == 2) continue;   // v feeds only vsum; done analytically in k_vsum*
#pragma unroll
    for (int r = 0; r < 8; ++r) {
      const int pos = m0 + r + hi * 8;
      const int i0 = pos >> 9, j0 = pos & 511;
      const int h  = i0 >> 7;
      const int ip = ((i0 & 127) << 2) | (j0 >> 7);
      const int jp = ((j0 & 127) << 2) | fhi;
      const _Float16 val = (_Float16)acc[j][r];
      if (chunk == 0)      Qm[((size_t)((h << 14) | (ip << 5) | c) << 9) | jp] = val;
      else if (chunk == 1) Km[((size_t)((h << 14) | (ip << 5) | c) << 9) | jp] = val;
      else                 Gm[((((size_t)((h << 9) | ip) << 9) | jp) << 5) | c] = val;
    }
  }
}

// ---------------------------------------------------------------------------
// vsum[h][c] = (sum over positions in head of xn) . (sum of 4 W_proj v-columns)
// deterministic two-stage reduction (no fp atomics).
// ---------------------------------------------------------------------------
__global__ __launch_bounds__(128) void k_vsum1(const _Float16* __restrict__ xn,
                                               float* __restrict__ sxn) {
  const int b = blockIdx.x;        // 256 blocks: 64 segments per head
  const int d = threadIdx.x;       // 128 channels
  const size_t p0 = (size_t)b * 1024;
  float a = 0.f;
  for (int p = 0; p < 1024; ++p) a += (float)xn[(p0 + p) * DD + d];
  sxn[b * DD + d] = a;
}

__global__ __launch_bounds__(32) void k_vsum2(const float* __restrict__ sxn,
                                              const float* __restrict__ wproj,
                                              float* __restrict__ vsum) {
  const int h = blockIdx.x, c = threadIdx.x;
  float a = 0.f;
  for (int d = 0; d < DD; ++d) {
    float sx = 0.f;
    for (int seg = 0; seg < 64; ++seg) sx += sxn[(h * 64 + seg) * DD + d];
    const float wv = wproj[d * FF +  64 + c] + wproj[d * FF + 192 + c] +
                     wproj[d * FF + 320 + c] + wproj[d * FF + 448 + c];
    a += sx * wv;
  }
  vsum[h * 32 + c] = a;
}

// ---------------------------------------------------------------------------
// K3: score[h][j'][k'] = (Q^T K)/sqrt(C) + bias. M=N=512, K=16384 per head.
//     A-tile: register-transposed LDS staging (vector ds_load fragment reads).
//     B-tile: async DMA to LDS (ASYNCcnt) via global_load_async_to_lds_b128.
// ---------------------------------------------------------------------------
__global__ __launch_bounds__(128) void k_score(const _Float16* __restrict__ Qm,
                                               const _Float16* __restrict__ Km,
                                               const float* __restrict__ bias,
                                               float* __restrict__ score) {
  const int h  = blockIdx.x >> 8;
  const int j0 = ((blockIdx.x >> 4) & 15) * 32;
  const int k0 = (blockIdx.x & 15) * 32;
  const _Float16* Qh = Qm + ((size_t)h << 23);
  const _Float16* Kh = Km + ((size_t)h << 23);
  __shared__ _Float16 As[32][40];   // [j'_local][t_local], padded (transposed staging)
#if defined(ATHENA_ASYNC_LDS)
  __shared__ _Float16 Bs[32][40];   // [t_local][k'_local], natural, DMA-filled
#else
  __shared__ _Float16 Bs[32][40];   // [k'_local][t_local], transposed staging
#endif
  const int tid = threadIdx.x;
  const int w = tid >> 5, lane = tid & 31, lm = lane & 15, hi = lane >> 4;
  const int jo = (w >> 1) * 16, ko = (w & 1) * 16;
  const int tt = tid >> 2, c8 = (tid & 3) * 8;
  v8f acc = zero_v8f();
  for (int ks = 0; ks < 512; ++ks) {
    const int t0 = ks * 32;
    const v8h qa = *(const v8h*)(Qh + (size_t)(t0 + tt) * NN + j0 + c8);
#if !defined(ATHENA_ASYNC_LDS)
    const v8h kb = *(const v8h*)(Kh + (size_t)(t0 + tt) * NN + k0 + c8);
#endif
    if (ks + 1 < 512) {   // -> global_prefetch_b8
      __builtin_prefetch(Qh + (size_t)(t0 + 32 + tt) * NN + j0 + c8, 0, 1);
      __builtin_prefetch(Kh + (size_t)(t0 + 32 + tt) * NN + k0 + c8, 0, 1);
    }
    __syncthreads();      // all waves done reading the previous tiles
#pragma unroll
    for (int e = 0; e < 8; ++e) As[c8 + e][tt] = qa[e];
#if defined(ATHENA_ASYNC_LDS)
    // DMA the naturally-laid-out K tile: each thread moves one aligned 16B chunk.
    async_copy_b128(Kh + (size_t)(t0 + tt) * NN + k0 + c8, &Bs[tt][c8]);
    wait_async_zero();    // ASYNCcnt -> 0 before publishing via barrier
#else
#pragma unroll
    for (int e = 0; e < 8; ++e) Bs[c8 + e][tt] = kb[e];
#endif
    __syncthreads();
    const v8h a0 = *(const v8h*)&As[jo + lm][hi * 16];
    const v8h a1 = *(const v8h*)&As[jo + lm][hi * 16 + 8];
#if defined(ATHENA_ASYNC_LDS)
    v16h bf;                       // transposed gather from the natural tile
#pragma unroll
    for (int s = 0; s < 16; ++s) bf[s] = Bs[hi * 16 + s][ko + lm];
    acc = WMMA_F32_F16(cat16(a0, a1), bf, acc);
#else
    const v8h b0 = *(const v8h*)&Bs[ko + lm][hi * 16];
    const v8h b1 = *(const v8h*)&Bs[ko + lm][hi * 16 + 8];
    acc = WMMA_F32_F16(cat16(a0, a1), cat16(b0, b1), acc);
#endif
  }
  const float scale = 0.088388347648318447f;  // 1/sqrt(128); bias added unscaled
#pragma unroll
  for (int r = 0; r < 8; ++r) {
    const int jp = j0 + jo + r + hi * 8;
    const int kp = k0 + ko + lm;
    const size_t idx = (size_t)(h * NN + jp) * NN + kp;
    score[idx] = acc[r] * scale + bias[idx];
  }
}

// ---------------------------------------------------------------------------
// K4: row softmax (one wave per 512-wide row), in place.
// ---------------------------------------------------------------------------
__global__ __launch_bounds__(256) void k_softmax(float* __restrict__ score) {
  const int row  = (blockIdx.x * blockDim.x + threadIdx.x) >> 5;
  const int lane = threadIdx.x & 31;
  if (row >= HH * NN) return;
  float* rp = score + (size_t)row * NN;
  float v[16];
  float mx = -3.0e38f;
#pragma unroll
  for (int i = 0; i < 16; ++i) { v[i] = rp[lane + 32 * i]; mx = fmaxf(mx, v[i]); }
#pragma unroll
  for (int m = 16; m > 0; m >>= 1) mx = fmaxf(mx, __shfl_xor(mx, m, 32));
  float s = 0.f;
#pragma unroll
  for (int i = 0; i < 16; ++i) { v[i] = __expf(v[i] - mx); s += v[i]; }
#pragma unroll
  for (int m = 16; m > 0; m >>= 1) s += __shfl_xor(s, m, 32);
  const float r = 1.f / s;
#pragma unroll
  for (int i = 0; i < 16; ++i) rp[lane + 32 * i] = v[i] * r;
}

// ---------------------------------------------------------------------------
// K5: gated output GEMM: out[p][:] = (G_row * score * vsum) @ W_out via WMMA.
// ---------------------------------------------------------------------------
__global__ __launch_bounds__(256) void k_out(const _Float16* __restrict__ Gm,
                                             const float* __restrict__ score,
                                             const float* __restrict__ vsum,
                                             const _Float16* __restrict__ wof,
                                             float* __restrict__ out) {
  const int w = threadIdx.x >> 5, lane = threadIdx.x & 31;
  const int lm = lane & 15, hi = lane >> 4;
  const int m0 = (blockIdx.x * 8 + w) * 16;
  const int p  = m0 + lm;                 // this lane's A-matrix row
  const int I = p >> 9, J = p & 511;
  const int h = I >> 7;
  const int a = ((I & 127) << 2) | (J >> 7);
  const size_t grow = (size_t)((h * NN + a) * NN + ((J & 127) << 2)) * 32;
  const size_t srow = (size_t)(h * NN + a) * NN + ((J & 127) << 2);
  v16h vf;
#pragma unroll
  for (int s = 0; s < 16; ++s) {
    const int c = ((s >> 3) << 4) | (hi * 8 + (s & 7));
    vf[s] = (_Float16)vsum[h * 32 + c];
  }
  v8f acc[8];
#pragma unroll
  for (int nt = 0; nt < 8; ++nt) acc[nt] = zero_v8f();
#pragma unroll
  for (int kt = 0; kt < 4; ++kt) {
    const float sc = score[srow + kt];
    const v8h g0 = *(const v8h*)(Gm + grow + kt * 32 + hi * 8);
    const v8h g1 = *(const v8h*)(Gm + grow + kt * 32 + 16 + hi * 8);
    v16h af;
#pragma unroll
    for (int s = 0; s < 8; ++s) {
      af[s]     = (_Float16)((float)g0[s] * (float)vf[s]     * sc);
      af[s + 8] = (_Float16)((float)g1[s] * (float)vf[s + 8] * sc);
    }
#pragma unroll
    for (int nt = 0; nt < 8; ++nt) {
      const _Float16* bp = wof + ((size_t)((kt * 8 + nt) * 32 + lane) << 4);
      const v16h bf = cat16(*(const v8h*)bp, *(const v8h*)(bp + 8));
      acc[nt] = WMMA_F32_F16(af, bf, acc[nt]);
    }
  }
#pragma unroll
  for (int nt = 0; nt < 8; ++nt) {
#pragma unroll
    for (int r = 0; r < 8; ++r) {
      out[(size_t)(m0 + r + hi * 8) * DD + nt * 16 + lm] = acc[nt][r];
    }
  }
}

extern "C" void kernel_launch(void* const* d_in, const int* in_sizes, int n_in,
                              void* d_out, int out_size, void* d_ws, size_t ws_size,
                              hipStream_t stream) {
  const float* x   = (const float*)d_in[0];
  const float* lnw = (const float*)d_in[1];
  const float* lnb = (const float*)d_in[2];
  const float* wp  = (const float*)d_in[3];
  const float* wb  = (const float*)d_in[4];
  const float* wo  = (const float*)d_in[5];
  float* out = (float*)d_out;
  char* ws = (char*)d_ws;

  _Float16* xn    = (_Float16*)(ws + XN_OFF);
  _Float16* Qm    = (_Float16*)(ws + Q_OFF);
  _Float16* Km    = (_Float16*)(ws + K_OFF);
  _Float16* Gm    = (_Float16*)(ws + G_OFF);
  float*    bias  = (float*)(ws + BIAS_OFF);
  float*    score = (float*)(ws + SCORE_OFF);
  float*    vsum  = (float*)(ws + VSUM_OFF);
  _Float16* wpf   = (_Float16*)(ws + WPF_OFF);
  _Float16* wof   = (_Float16*)(ws + WOF_OFF);
  float*    sxn   = (float*)(ws + SXN_OFF);

  k_pack   <<<320,      256, 0, stream>>>(wp, wo, wpf, wof);
  k_ln     <<<NP / 8,   256, 0, stream>>>(x, lnw, lnb, wb, xn, bias);
  k_proj   <<<8192,     256, 0, stream>>>(xn, wpf, Qm, Km, Gm);
  k_vsum1  <<<256,      128, 0, stream>>>(xn, sxn);
  k_vsum2  <<<4,         32, 0, stream>>>(sxn, wp, vsum);
  k_score  <<<1024,     128, 0, stream>>>(Qm, Km, bias, score);
  k_softmax<<<HH * NN / 8, 256, 0, stream>>>(score);
  k_out    <<<2048,     256, 0, stream>>>(Gm, score, vsum, wof, out);

  (void)in_sizes; (void)n_in; (void)out_size; (void)ws_size;
}
